// Free_Block_54803782697338
// MI455X (gfx1250) — matrix operand order repaired and checked by reading, not compiled, not run on hardware
//
#include <hip/hip_runtime.h>

// Problem constants (match reference)
#define B_  64
#define C_  4
#define T_  256
#define I_  1024
#define O_  256
#define K_  8

typedef __attribute__((ext_vector_type(16))) __bf16 v16bf;
typedef __attribute__((ext_vector_type(8)))  __bf16 v8bf;
typedef __attribute__((ext_vector_type(8)))  float  v8f;

union V16U { v16bf v; v8bf h[2]; };

// f32 -> bf16 with round-to-nearest-even (pure bit ops; no libcalls)
__device__ __forceinline__ __bf16 f2bf(float f) {
  union { float f; unsigned u; } in; in.f = f;
  unsigned r = (in.u + 0x7FFFu + ((in.u >> 16) & 1u)) >> 16;
  union { unsigned short s; __bf16 b; } out; out.s = (unsigned short)r;
  return out.b;
}

// ---- CDNA5 async global->LDS copy (ASYNCcnt-tracked), with fallback ------
#if __has_builtin(__builtin_amdgcn_global_load_async_to_lds_b128)
#define HAS_ASYNC_LDS 1
// exact param flavor per hipcc diagnostic: vector_size(16) int, AS(1)/AS(3)
typedef int int4v __attribute__((vector_size(16)));
typedef __attribute__((address_space(1))) int4v gint4;
typedef __attribute__((address_space(3))) int4v lint4;
__device__ __forceinline__ void async_copy16(const void* g, void* l) {
  __builtin_amdgcn_global_load_async_to_lds_b128(
      (gint4*)(int4v*)(void*)g, (lint4*)(int4v*)l, 0, 0);
}
__device__ __forceinline__ void async_wait0() {
#if __has_builtin(__builtin_amdgcn_s_wait_asynccnt)
  __builtin_amdgcn_s_wait_asynccnt(0);
#else
  asm volatile("s_wait_asynccnt 0x0" ::: "memory");
#endif
}
#else
#define HAS_ASYNC_LDS 0
__device__ __forceinline__ void async_copy16(const void* g, void* l) {
  *(v8bf*)l = *(const v8bf*)g;   // fallback: through VGPRs
}
__device__ __forceinline__ void async_wait0() {}
#endif

// ---------------------------------------------------------------------------
// Kernel 1: pre-merge x with wm:  xm[t][b][i] = sum_c wm[t,c] * x[b,c,i]  (bf16)
// grid (B, I/256), block 256
// ---------------------------------------------------------------------------
__global__ __launch_bounds__(256) void xm_kernel(
    const float* __restrict__ x, const float* __restrict__ wm,
    __bf16* __restrict__ xm)
{
  __shared__ float wms[T_ * C_];
  const int tid = threadIdx.x;
  for (int idx = tid; idx < T_ * C_; idx += 256) wms[idx] = wm[idx];
  __syncthreads();

  const int b = blockIdx.x;
  const int i = blockIdx.y * 256 + tid;
  const float x0 = x[(size_t)b * C_ * I_ + 0 * I_ + i];
  const float x1 = x[(size_t)b * C_ * I_ + 1 * I_ + i];
  const float x2 = x[(size_t)b * C_ * I_ + 2 * I_ + i];
  const float x3 = x[(size_t)b * C_ * I_ + 3 * I_ + i];

  for (int t = 0; t < T_; ++t) {
    float v = wms[t*4+0]*x0 + wms[t*4+1]*x1 + wms[t*4+2]*x2 + wms[t*4+3]*x3;
    xm[(size_t)t * (B_ * I_) + (size_t)b * I_ + i] = f2bf(v);
  }
}

// ---------------------------------------------------------------------------
// Kernel 2: per-token fused pipeline (one block per token t):
//   tokens = relu(Xm_t[64x1024] * Wg_t^T + s_t*bg_t)
//   hdn    = relu(tokens * W1_t^T + b1_t)
//   proj   = hdn * W2_t^T + b2_t
// 8 wave32s: wave w -> row strip (w>>1)*16, col half (w&1)*128 (8 16x16 tiles)
// ---------------------------------------------------------------------------
__global__ __launch_bounds__(256) void token_pipeline_kernel(
    const float* __restrict__ Wg, const float* __restrict__ bg,
    const float* __restrict__ wm, const float* __restrict__ W1,
    const float* __restrict__ b1, const float* __restrict__ W2,
    const float* __restrict__ b2, const __bf16* __restrict__ xm,
    float* __restrict__ out_tok, float* __restrict__ out_proj)
{
  __shared__ __bf16 atile[64][32];    // A tile (async-staged xm)     (4 KB)
  __shared__ __bf16 btile[256][32];   // B tile, transposed [n][k]   (16 KB)
  __shared__ __bf16 act0[64][256];    // tokens (bf16)               (32 KB)
  __shared__ __bf16 act1[64][256];    // hdn    (bf16)               (32 KB)

  const int t    = blockIdx.x;
  const int tid  = threadIdx.x;
  const int wave = tid >> 5;
  const int lane = tid & 31;
  const int rs   = wave >> 1;        // row strip 0..3
  const int ch   = wave & 1;         // column half 0..1
  const int lm   = lane & 15;
  const int lh   = lane >> 4;
  const int row  = rs * 16 + lm;     // A-matrix row this lane owns

  const float s_t = wm[t*4+0] + wm[t*4+1] + wm[t*4+2] + wm[t*4+3];

  v8f acc[8];

  // ======================= Phase 1: tokens ================================
  #pragma unroll
  for (int ct = 0; ct < 8; ++ct)
    #pragma unroll
    for (int r = 0; r < 8; ++r) acc[ct][r] = 0.0f;

  const float*  wgt = Wg + (size_t)t * O_ * I_;   // [o][i]
  const __bf16* xmt = xm + (size_t)t * B_ * I_;   // [b][i]

  // async A-tile copy slot for this thread: row tid>>2, 8 bf16 at (tid&3)*8
  const int arow = tid >> 2;
  const int aoff = (tid & 3) * 8;

  for (int k0 = 0; k0 < I_; k0 += 32) {
    __syncthreads();
    // kick off async copy of the 64x32 bf16 A tile into LDS (ASYNCcnt)
    async_copy16(xmt + (size_t)arow * I_ + k0 + aoff, &atile[arow][aoff]);
    {   // stage B tile: thread tid = column n, convert f32 -> bf16
      const float* src = wgt + (size_t)tid * I_ + k0;
      __builtin_prefetch(src + 32, 0, 1);
      #pragma unroll
      for (int j = 0; j < 32; j += 4) {
        float4 f = *(const float4*)(src + j);
        btile[tid][j+0] = f2bf(f.x);
        btile[tid][j+1] = f2bf(f.y);
        btile[tid][j+2] = f2bf(f.z);
        btile[tid][j+3] = f2bf(f.w);
      }
    }
    async_wait0();
    __syncthreads();

    V16U a;   // 16x32 bf16 A-frag: elems 0-7 K=lh*8.., elems 8-15 K=16+lh*8..
    {
      const __bf16* ap = &atile[row][lh * 8];
      a.h[0] = *(const v8bf*)ap;
      a.h[1] = *(const v8bf*)(ap + 16);
    }
    #pragma unroll
    for (int ct = 0; ct < 8; ++ct) {
      const int n = ch * 128 + ct * 16 + lm;   // B col this lane owns
      V16U b;   // 32x16 bf16 B-frag: elems 0-7 K=lh*8.., elems 8-15 K=16+lh*8..
      const __bf16* bp = &btile[n][lh * 8];
      b.h[0] = *(const v8bf*)bp;
      b.h[1] = *(const v8bf*)(bp + 16);
      acc[ct] = __builtin_amdgcn_wmma_f32_16x16x32_bf16(
          false, a.v, false, b.v, (short)0, acc[ct], false, false);
    }
  }
  // epilogue: bias (merged through wm) + ReLU; keep bf16 copy in LDS
  #pragma unroll
  for (int ct = 0; ct < 8; ++ct) {
    const int col  = ch * 128 + ct * 16 + lm;
    const float bias = s_t * bg[t * O_ + col];
    #pragma unroll
    for (int r = 0; r < 8; ++r) {
      const int rr = rs * 16 + lh * 8 + r;     // C/D layout: lanes 16-31 -> M+8
      float v = acc[ct][r] + bias;
      v = v > 0.0f ? v : 0.0f;
      act0[rr][col] = f2bf(v);
      out_tok[(size_t)rr * (T_ * O_) + (size_t)t * O_ + col] = v;
    }
  }

  // ======================= Phase 2: hdn ===================================
  #pragma unroll
  for (int ct = 0; ct < 8; ++ct)
    #pragma unroll
    for (int r = 0; r < 8; ++r) acc[ct][r] = 0.0f;

  const float* w1t = W1 + (size_t)t * O_ * O_;  // [p][o]
  for (int k0 = 0; k0 < O_; k0 += 32) {
    __syncthreads();
    {
      const float* src = w1t + (size_t)tid * O_ + k0;
      __builtin_prefetch(src + 32, 0, 1);
      #pragma unroll
      for (int j = 0; j < 32; j += 4) {
        float4 f = *(const float4*)(src + j);
        btile[tid][j+0] = f2bf(f.x);
        btile[tid][j+1] = f2bf(f.y);
        btile[tid][j+2] = f2bf(f.z);
        btile[tid][j+3] = f2bf(f.w);
      }
    }
    __syncthreads();

    V16U a;
    {
      const __bf16* ap = &act0[row][k0 + lh * 8];
      a.h[0] = *(const v8bf*)ap;
      a.h[1] = *(const v8bf*)(ap + 16);
    }
    #pragma unroll
    for (int ct = 0; ct < 8; ++ct) {
      const int n = ch * 128 + ct * 16 + lm;
      V16U b;
      const __bf16* bp = &btile[n][lh * 8];
      b.h[0] = *(const v8bf*)bp;
      b.h[1] = *(const v8bf*)(bp + 16);
      acc[ct] = __builtin_amdgcn_wmma_f32_16x16x32_bf16(
          false, a.v, false, b.v, (short)0, acc[ct], false, false);
    }
  }
  #pragma unroll
  for (int ct = 0; ct < 8; ++ct) {
    const int col  = ch * 128 + ct * 16 + lm;
    const float bias = b1[t * O_ + col];
    #pragma unroll
    for (int r = 0; r < 8; ++r) {
      const int rr = rs * 16 + lh * 8 + r;
      float v = acc[ct][r] + bias;
      v = v > 0.0f ? v : 0.0f;
      act1[rr][col] = f2bf(v);
    }
  }

  // ======================= Phase 3: proj ==================================
  #pragma unroll
  for (int ct = 0; ct < 8; ++ct)
    #pragma unroll
    for (int r = 0; r < 8; ++r) acc[ct][r] = 0.0f;

  const float* w2t = W2 + (size_t)t * O_ * O_;  // [o][p]
  for (int k0 = 0; k0 < O_; k0 += 32) {
    __syncthreads();
    {
      const float* src = w2t + (size_t)tid * O_ + k0;
      __builtin_prefetch(src + 32, 0, 1);
      #pragma unroll
      for (int j = 0; j < 32; j += 4) {
        float4 f = *(const float4*)(src + j);
        btile[tid][j+0] = f2bf(f.x);
        btile[tid][j+1] = f2bf(f.y);
        btile[tid][j+2] = f2bf(f.z);
        btile[tid][j+3] = f2bf(f.w);
      }
    }
    __syncthreads();

    V16U a;
    {
      const __bf16* ap = &act1[row][k0 + lh * 8];
      a.h[0] = *(const v8bf*)ap;
      a.h[1] = *(const v8bf*)(ap + 16);
    }
    #pragma unroll
    for (int ct = 0; ct < 8; ++ct) {
      const int n = ch * 128 + ct * 16 + lm;
      V16U b;
      const __bf16* bp = &btile[n][lh * 8];
      b.h[0] = *(const v8bf*)bp;
      b.h[1] = *(const v8bf*)(bp + 16);
      acc[ct] = __builtin_amdgcn_wmma_f32_16x16x32_bf16(
          false, a.v, false, b.v, (short)0, acc[ct], false, false);
    }
  }
  #pragma unroll
  for (int ct = 0; ct < 8; ++ct) {
    const int col  = ch * 128 + ct * 16 + lm;
    const float bias = b2[t * O_ + col];
    #pragma unroll
    for (int r = 0; r < 8; ++r) {
      const int rr = rs * 16 + lh * 8 + r;
      out_proj[(size_t)rr * (T_ * O_) + (size_t)t * O_ + col] = acc[ct][r] + bias;
    }
  }
}

// ---------------------------------------------------------------------------
// Kernel 3: aggregator  out_img[b][k][o] = sum_t proj[b][t][o] * Wa[k][t]
// grid B, block 256 (thread = o)
// ---------------------------------------------------------------------------
__global__ __launch_bounds__(256) void agg_kernel(
    const float* __restrict__ proj, const float* __restrict__ Wa,
    float* __restrict__ out_img)
{
  __shared__ float was[K_ * T_];
  const int tid = threadIdx.x;
  for (int idx = tid; idx < K_ * T_; idx += 256) was[idx] = Wa[idx];
  __syncthreads();

  const int b = blockIdx.x;
  const int o = tid;
  float acc[K_];
  #pragma unroll
  for (int k = 0; k < K_; ++k) acc[k] = 0.0f;

  const float* pb = proj + (size_t)b * (T_ * O_) + o;
  for (int t = 0; t < T_; ++t) {
    float p = pb[(size_t)t * O_];
    #pragma unroll
    for (int k = 0; k < K_; ++k) acc[k] += was[k * T_ + t] * p;
  }
  #pragma unroll
  for (int k = 0; k < K_; ++k)
    out_img[(size_t)b * (K_ * O_) + k * O_ + o] = acc[k];
}

// ---------------------------------------------------------------------------
extern "C" void kernel_launch(void* const* d_in, const int* in_sizes, int n_in,
                              void* d_out, int out_size, void* d_ws, size_t ws_size,
                              hipStream_t stream) {
  const float* x  = (const float*)d_in[0];
  const float* Wg = (const float*)d_in[1];
  const float* bg = (const float*)d_in[2];
  const float* wm = (const float*)d_in[3];
  const float* W1 = (const float*)d_in[4];
  const float* b1 = (const float*)d_in[5];
  const float* W2 = (const float*)d_in[6];
  const float* b2 = (const float*)d_in[7];
  const float* Wa = (const float*)d_in[8];

  float* out      = (float*)d_out;
  float* out_img  = out;                               // B*K*OS*OS = 131072
  float* out_tok  = out + (size_t)B_ * K_ * O_;        // B*T*O
  float* out_proj = out_tok + (size_t)B_ * T_ * O_;    // B*T*O

  __bf16* xm = (__bf16*)d_ws;                          // T*B*I bf16 = 32 MB

  xm_kernel<<<dim3(B_, I_ / 256), dim3(256), 0, stream>>>(x, wm, xm);
  token_pipeline_kernel<<<dim3(T_), dim3(256), 0, stream>>>(
      Wg, bg, wm, W1, b1, W2, b2, xm, out_tok, out_proj);
  agg_kernel<<<dim3(B_), dim3(256), 0, stream>>>(out_proj, Wa, out_img);
}
